// Decoder_27874337751622
// MI455X (gfx1250) — matrix operand order repaired and checked by reading, not compiled, not run on hardware
//
#include <hip/hip_runtime.h>
#include <hip/hip_bf16.h>

typedef __bf16 bf16_t;
typedef __attribute__((ext_vector_type(16))) __bf16 v16bf;
typedef __attribute__((ext_vector_type(8)))  __bf16 v8bf;
typedef __attribute__((ext_vector_type(8)))  float  v8f;

#define L_  6
#define H_  12
#define D_  768
#define DH_ 64
#define F_  3072
#define V_  32000
#define S_  1024
#define B_  2
#define M_  (B_ * S_)

// ---- CDNA5 async global->LDS copy (ASYNCcnt), with safe fallback ----------
#if __has_builtin(__builtin_amdgcn_global_load_async_to_lds_b128) && \
    __has_builtin(__builtin_amdgcn_s_wait_asynccnt)
#define HAVE_ASYNC_LDS 1
#else
#define HAVE_ASYNC_LDS 0
#endif

typedef int v4i_ __attribute__((vector_size(16)));
typedef v4i_ __attribute__((address_space(3)))* lds_v4i_ptr;

__device__ __forceinline__ void copy16B_to_lds(const bf16_t* g, bf16_t* l) {
#if HAVE_ASYNC_LDS
    // param0: generic int4* (per clang diagnostic); param1: LDS int4*
    __builtin_amdgcn_global_load_async_to_lds_b128(
        (v4i_*)(bf16_t*)g, (lds_v4i_ptr)l, 0, 0);
#else
    *(uint4*)l = *(const uint4*)g;
#endif
}
__device__ __forceinline__ void wait_lds_copies() {
#if HAVE_ASYNC_LDS
    __builtin_amdgcn_s_wait_asynccnt(0);
#endif
}

// Concatenate two 8-element bf16 vectors into a 16-element fragment.
__device__ __forceinline__ v16bf concat8(v8bf lo, v8bf hi) {
    return __builtin_shufflevector(lo, hi, 0,1,2,3,4,5,6,7,8,9,10,11,12,13,14,15);
}
__device__ __forceinline__ v8f wmma_bf16(v16bf a, v16bf b, v8f c) {
    return __builtin_amdgcn_wmma_f32_16x16x32_bf16(
        false, a, false, b, (short)0, c, false, false);
}
// A-frag (16x32): per lane contiguous 16 elems at k = 16h.
__device__ __forceinline__ v16bf load_a_frag(const bf16_t* base) {
    return concat8(*(const v8bf*)(base), *(const v8bf*)(base + 8));
}
// B-frag (32x16, stored [n][k]): per lane contiguous 16 elems at k = 16h.
__device__ __forceinline__ v16bf load_b_frag(const bf16_t* base) {
    return concat8(*(const v8bf*)(base), *(const v8bf*)(base + 8));
}

// ================= Weight packing: fp32 -> bf16, K-tiled [K/32][N][32] =====
__global__ __launch_bounds__(256) void pack_tiled_kernel(
    const float* __restrict__ src, bf16_t* __restrict__ dst, int K, int N, int nl)
{
    size_t per = (size_t)K * N;
    size_t n_tot = per * nl;
    size_t i = (size_t)blockIdx.x * 256u + threadIdx.x;
    size_t stride = (size_t)gridDim.x * 256u;
    for (; i < n_tot; i += stride) {
        size_t l = i / per, r = i % per;
        size_t kt = r / ((size_t)N * 32), r2 = r % ((size_t)N * 32);
        size_t n = r2 / 32, kk = r2 % 32;
        size_t k = kt * 32 + kk;
        dst[i] = (bf16_t)src[l * per + k * N + n];
    }
}

// src [L,H,D,DH] -> tiled per layer with K=D, N=H*DH (n = h*64+dh)
__global__ __launch_bounds__(256) void pack_qkv_kernel(
    const float* __restrict__ src, bf16_t* __restrict__ dst)
{
    size_t per = (size_t)D_ * D_;
    size_t n_tot = per * L_;
    size_t i = (size_t)blockIdx.x * 256u + threadIdx.x;
    size_t stride = (size_t)gridDim.x * 256u;
    for (; i < n_tot; i += stride) {
        size_t l = i / per, r = i % per;
        size_t kt = r / ((size_t)D_ * 32), r2 = r % ((size_t)D_ * 32);
        size_t n = r2 / 32, kk = r2 % 32;
        size_t d = kt * 32 + kk;
        size_t h = n >> 6, dh = n & 63;
        dst[i] = (bf16_t)src[(((l * H_ + h) * D_) + d) * DH_ + dh];
    }
}

// emb [V,D] -> tiled embT with K=D, N=V (reads coalesced in kk==d)
__global__ __launch_bounds__(256) void pack_embT_kernel(
    const float* __restrict__ src, bf16_t* __restrict__ dst)
{
    size_t n_tot = (size_t)D_ * V_;
    size_t i = (size_t)blockIdx.x * 256u + threadIdx.x;
    size_t stride = (size_t)gridDim.x * 256u;
    for (; i < n_tot; i += stride) {
        size_t kt = i / ((size_t)V_ * 32), r2 = i % ((size_t)V_ * 32);
        size_t n = r2 / 32, kk = r2 % 32;
        size_t d = kt * 32 + kk;
        dst[i] = (bf16_t)src[n * D_ + d];
    }
}

// ================= Embedding lookup + positional add =======================
__global__ __launch_bounds__(256) void embed_kernel(
    const int* __restrict__ X, const float* __restrict__ emb,
    const float* __restrict__ pos, float* __restrict__ x)
{
    int row = blockIdx.x;
    int s   = row & (S_ - 1);
    int tok = X[row];
    for (int i = threadIdx.x; i < D_; i += 256)
        x[(size_t)row * D_ + i] = emb[(size_t)tok * D_ + i] + pos[(size_t)s * D_ + i];
}

// ================= LayerNorm (fp32 in, bf16 out) ===========================
__global__ __launch_bounds__(256) void layernorm_kernel(
    const float* __restrict__ x, const float* __restrict__ sc,
    const float* __restrict__ bi, bf16_t* __restrict__ out)
{
    __shared__ float red1[256];
    __shared__ float red2[256];
    int row = blockIdx.x;
    int t = threadIdx.x;
    const float* xr = x + (size_t)row * D_;
    float a0 = xr[t], a1 = xr[t + 256], a2 = xr[t + 512];
    red1[t] = a0 + a1 + a2;
    red2[t] = a0 * a0 + a1 * a1 + a2 * a2;
    __syncthreads();
    for (int off = 128; off > 0; off >>= 1) {
        if (t < off) { red1[t] += red1[t + off]; red2[t] += red2[t + off]; }
        __syncthreads();
    }
    float mu   = red1[0] * (1.0f / D_);
    float var  = red2[0] * (1.0f / D_) - mu * mu;
    float rstd = rsqrtf(var + 1e-5f);
    size_t ob = (size_t)row * D_;
    out[ob + t]       = (bf16_t)((a0 - mu) * rstd * sc[t]       + bi[t]);
    out[ob + t + 256] = (bf16_t)((a1 - mu) * rstd * sc[t + 256] + bi[t + 256]);
    out[ob + t + 512] = (bf16_t)((a2 - mu) * rstd * sc[t + 512] + bi[t + 512]);
}

// ================= bf16 WMMA GEMM (128x128 block, 8 waves) =================
// A [M,K] row-major bf16; B tiled [K/32][N][32] bf16.
#define GM 128
#define GN 128
#define GK 32
#define LDS_STRIDE 48   // 96B rows; every frag read is 16B aligned

enum { MODE_BF16_BIAS = 0, MODE_BF16_RELU = 1, MODE_QKV = 2,
       MODE_RES_F32 = 3, MODE_F32_BIAS = 4, MODE_VT = 5 };

__global__ __launch_bounds__(256) void gemm_bf16_kernel(
    const bf16_t* __restrict__ A, const bf16_t* __restrict__ B,
    const float* __restrict__ bias, void* __restrict__ out,
    int M, int N, int K, int mode)
{
    __shared__ bf16_t sA[GM * LDS_STRIDE];   // [m][k]
    __shared__ bf16_t sBt[GN * LDS_STRIDE];  // [n][k]

    int tid  = threadIdx.x;
    int lane = tid & 31;
    int wave = tid >> 5;
    int wrow = (wave & 3) * 32;      // wave's 32-row span within 128
    int wcol = (wave >> 2) * 64;     // wave's 64-col span within 128
    int m0 = blockIdx.y * GM;
    int n0 = blockIdx.x * GN;

    int half = (lane >> 4) & 1;
    int ncol = lane & 15;

    // tile copy mapping: 256 threads x 16 bf16 (2 x b128) cover 128x32
    int ldRow = tid >> 1, ldCol = (tid & 1) * 16;
    const bf16_t* gA = A + (size_t)(m0 + ldRow) * K + ldCol;
    const bf16_t* gB = B + ((size_t)(n0 + ldRow)) * 32 + ldCol;
    bf16_t* lA = sA  + ldRow * LDS_STRIDE + ldCol;
    bf16_t* lB = sBt + ldRow * LDS_STRIDE + ldCol;

    // fragment read bases
    const bf16_t* aB0 = sA + (wrow + ncol) * LDS_STRIDE + half * 16;
    const bf16_t* aB1 = aB0 + 16 * LDS_STRIDE;
    const bf16_t* bB0 = sBt + (wcol + ncol) * LDS_STRIDE + half * 16;

    v8f c[2][4];
    #pragma unroll
    for (int rt = 0; rt < 2; ++rt)
        #pragma unroll
        for (int ct = 0; ct < 4; ++ct) c[rt][ct] = (v8f){};

    for (int k0 = 0; k0 < K; k0 += GK, gA += GK, gB += (size_t)N * 32) {
        copy16B_to_lds(gA,     lA);
        copy16B_to_lds(gA + 8, lA + 8);
        copy16B_to_lds(gB,     lB);
        copy16B_to_lds(gB + 8, lB + 8);
        wait_lds_copies();
        __syncthreads();

        v16bf a0 = load_a_frag(aB0);
        v16bf a1 = load_a_frag(aB1);
        #pragma unroll
        for (int ct = 0; ct < 4; ++ct) {
            v16bf bf = load_b_frag(bB0 + ct * 16 * LDS_STRIDE);
            c[0][ct] = wmma_bf16(a0, bf, c[0][ct]);
            c[1][ct] = wmma_bf16(a1, bf, c[1][ct]);
        }
        __syncthreads();
    }

    #pragma unroll
    for (int rt = 0; rt < 2; ++rt) {
        #pragma unroll
        for (int ct = 0; ct < 4; ++ct) {
            v8f cc = c[rt][ct];
            int n = n0 + wcol + ct * 16 + ncol;
            float bn = bias[n];
            #pragma unroll
            for (int i = 0; i < 8; ++i) {
                int m = m0 + wrow + rt * 16 + i + half * 8;
                float v = cc[i] + bn;
                if (mode == MODE_BF16_BIAS) {
                    ((bf16_t*)out)[(size_t)m * N + n] = (bf16_t)v;
                } else if (mode == MODE_BF16_RELU) {
                    ((bf16_t*)out)[(size_t)m * N + n] = (bf16_t)fmaxf(v, 0.0f);
                } else if (mode == MODE_QKV) {
                    int b = m >> 10, s = m & (S_ - 1);
                    ((bf16_t*)out)[(((size_t)b * H_ + (n >> 6)) * S_ + s) * DH_ + (n & 63)]
                        = (bf16_t)v;
                } else if (mode == MODE_VT) {
                    int b = m >> 10, s = m & (S_ - 1);
                    ((bf16_t*)out)[(((size_t)b * H_ + (n >> 6)) * DH_ + (n & 63)) * S_ + s]
                        = (bf16_t)v;
                } else if (mode == MODE_RES_F32) {
                    ((float*)out)[(size_t)m * N + n] += v;
                } else { // MODE_F32_BIAS
                    ((float*)out)[(size_t)m * N + n] = v;
                }
            }
        }
    }
}

// ================= Flash-style causal attention ============================
// q,k: [B,H,S,DH] bf16; vT: [B,H,DH,S] bf16; out: [B,S,H*DH] bf16.
__global__ __launch_bounds__(32) void attn_kernel(
    const bf16_t* __restrict__ q, const bf16_t* __restrict__ k,
    const bf16_t* __restrict__ vT, bf16_t* __restrict__ out)
{
    int lane = threadIdx.x;
    int qb = blockIdx.x, h = blockIdx.y, b = blockIdx.z;
    size_t base = ((size_t)(b * H_ + h)) * S_ * DH_;
    const bf16_t* qp = q  + base;
    const bf16_t* kp = k  + base;
    const bf16_t* vp = vT + base;   // [DH][S]
    int q0   = qb * 16;
    int half = (lane >> 4) & 1;
    int ncol = lane & 15;
    const float scale = 0.125f;
    const float NEG_INF = -__builtin_inff();

    __shared__ bf16_t sP[16 * 32];

    const bf16_t* qrow = qp + (size_t)(q0 + ncol) * DH_;
    v16bf aq0 = concat8(*(const v8bf*)(qrow + half * 16),
                        *(const v8bf*)(qrow + 8 + half * 16));
    v16bf aq1 = concat8(*(const v8bf*)(qrow + 32 + half * 16),
                        *(const v8bf*)(qrow + 40 + half * 16));

    v8f o0 = {}, o1 = {}, o2 = {}, o3 = {};
    float mst[8], lst[8];
    #pragma unroll
    for (int i = 0; i < 8; ++i) { mst[i] = NEG_INF; lst[i] = 0.0f; }

    for (int kv0 = 0; kv0 <= q0 + 15; kv0 += 32) {
        v8f s0 = {}, s1 = {};
        {
            const bf16_t* kr0 = kp + (size_t)(kv0 + ncol) * DH_ + half * 16;
            const bf16_t* kr1 = kp + (size_t)(kv0 + 16 + ncol) * DH_ + half * 16;
            v16bf kf;
            kf = concat8(*(const v8bf*)(kr0), *(const v8bf*)(kr0 + 8));
            s0 = wmma_bf16(aq0, kf, s0);
            kf = concat8(*(const v8bf*)(kr0 + 32), *(const v8bf*)(kr0 + 40));
            s0 = wmma_bf16(aq1, kf, s0);
            kf = concat8(*(const v8bf*)(kr1), *(const v8bf*)(kr1 + 8));
            s1 = wmma_bf16(aq0, kf, s1);
            kf = concat8(*(const v8bf*)(kr1 + 32), *(const v8bf*)(kr1 + 40));
            s1 = wmma_bf16(aq1, kf, s1);
        }

        #pragma unroll
        for (int i = 0; i < 8; ++i) {
            int qr = q0 + i + half * 8;
            float sc0 = (kv0 + ncol      <= qr) ? s0[i] * scale : NEG_INF;
            float sc1 = (kv0 + 16 + ncol <= qr) ? s1[i] * scale : NEG_INF;
            float rm = fmaxf(sc0, sc1);
            rm = fmaxf(rm, __shfl_xor(rm, 1, 32));
            rm = fmaxf(rm, __shfl_xor(rm, 2, 32));
            rm = fmaxf(rm, __shfl_xor(rm, 4, 32));
            rm = fmaxf(rm, __shfl_xor(rm, 8, 32));
            float mnew = fmaxf(mst[i], rm);
            float corr = (mnew == NEG_INF) ? 1.0f : __expf(mst[i] - mnew);
            float p0 = (sc0 == NEG_INF) ? 0.0f : __expf(sc0 - mnew);
            float p1 = (sc1 == NEG_INF) ? 0.0f : __expf(sc1 - mnew);
            float ps = p0 + p1;
            ps += __shfl_xor(ps, 1, 32);
            ps += __shfl_xor(ps, 2, 32);
            ps += __shfl_xor(ps, 4, 32);
            ps += __shfl_xor(ps, 8, 32);
            lst[i] = lst[i] * corr + ps;
            mst[i] = mnew;
            o0[i] *= corr; o1[i] *= corr; o2[i] *= corr; o3[i] *= corr;
            sP[(i + half * 8) * 32 + ncol]      = (bf16_t)p0;
            sP[(i + half * 8) * 32 + 16 + ncol] = (bf16_t)p1;
        }
        __syncthreads();

        {
            const bf16_t* pr = sP + ncol * 32 + half * 16;
            v16bf ap = concat8(*(const v8bf*)(pr), *(const v8bf*)(pr + 8));
            const bf16_t* vr0 = vp + (size_t)(ncol)      * S_ + kv0 + half * 16;
            const bf16_t* vr1 = vp + (size_t)(ncol + 16) * S_ + kv0 + half * 16;
            const bf16_t* vr2 = vp + (size_t)(ncol + 32) * S_ + kv0 + half * 16;
            const bf16_t* vr3 = vp + (size_t)(ncol + 48) * S_ + kv0 + half * 16;
            v16bf vf;
            vf = concat8(*(const v8bf*)(vr0), *(const v8bf*)(vr0 + 8));
            o0 = wmma_bf16(ap, vf, o0);
            vf = concat8(*(const v8bf*)(vr1), *(const v8bf*)(vr1 + 8));
            o1 = wmma_bf16(ap, vf, o1);
            vf = concat8(*(const v8bf*)(vr2), *(const v8bf*)(vr2 + 8));
            o2 = wmma_bf16(ap, vf, o2);
            vf = concat8(*(const v8bf*)(vr3), *(const v8bf*)(vr3 + 8));
            o3 = wmma_bf16(ap, vf, o3);
        }
        __syncthreads();
    }

    #pragma unroll
    for (int i = 0; i < 8; ++i) {
        int r = q0 + i + half * 8;
        float inv = 1.0f / lst[i];
        size_t ob = ((size_t)(b * S_ + r)) * (H_ * DH_) + h * DH_;
        out[ob + ncol]      = (bf16_t)(o0[i] * inv);
        out[ob + 16 + ncol] = (bf16_t)(o1[i] * inv);
        out[ob + 32 + ncol] = (bf16_t)(o2[i] * inv);
        out[ob + 48 + ncol] = (bf16_t)(o3[i] * inv);
    }
}

// ===========================================================================
extern "C" void kernel_launch(void* const* d_in, const int* in_sizes, int n_in,
                              void* d_out, int out_size, void* d_ws, size_t ws_size,
                              hipStream_t stream) {
    (void)in_sizes; (void)n_in; (void)out_size; (void)ws_size;

    const int*   X      = (const int*)  d_in[0];
    const float* emb    = (const float*)d_in[1];
    const float* pos    = (const float*)d_in[2];
    const float* Wq     = (const float*)d_in[3];
    const float* bq     = (const float*)d_in[4];
    const float* Wk     = (const float*)d_in[5];
    const float* bk     = (const float*)d_in[6];
    const float* Wv     = (const float*)d_in[7];
    const float* bv     = (const float*)d_in[8];
    const float* Wo     = (const float*)d_in[9];
    const float* bo     = (const float*)d_in[10];
    const float* ln1_s  = (const float*)d_in[11];
    const float* ln1_b  = (const float*)d_in[12];
    const float* ln2_s  = (const float*)d_in[13];
    const float* ln2_b  = (const float*)d_in[14];
    const float* W1     = (const float*)d_in[15];
    const float* b1     = (const float*)d_in[16];
    const float* W2     = (const float*)d_in[17];
    const float* b2     = (const float*)d_in[18];
    const float* lnf_s  = (const float*)d_in[19];
    const float* lnf_b  = (const float*)d_in[20];
    const float* head_b = (const float*)d_in[21];

    size_t off = 0;
    auto alloc = [&](size_t bytes) -> void* {
        void* p = (char*)d_ws + off;
        off += (bytes + 255) & ~(size_t)255;
        return p;
    };
    bf16_t* embT = (bf16_t*)alloc(2ull * D_ * V_);
    bf16_t* WqP  = (bf16_t*)alloc(2ull * L_ * D_ * D_);
    bf16_t* WkP  = (bf16_t*)alloc(2ull * L_ * D_ * D_);
    bf16_t* WvP  = (bf16_t*)alloc(2ull * L_ * D_ * D_);
    bf16_t* WoP  = (bf16_t*)alloc(2ull * L_ * D_ * D_);
    bf16_t* W1P  = (bf16_t*)alloc(2ull * L_ * D_ * F_);
    bf16_t* W2P  = (bf16_t*)alloc(2ull * L_ * F_ * D_);
    float*  xbuf = (float*) alloc(4ull * M_ * D_);
    bf16_t* xn   = (bf16_t*)alloc(2ull * M_ * D_);
    bf16_t* qb_  = (bf16_t*)alloc(2ull * M_ * D_);
    bf16_t* kb_  = (bf16_t*)alloc(2ull * M_ * D_);
    bf16_t* vb_  = (bf16_t*)alloc(2ull * M_ * D_);
    bf16_t* atto = (bf16_t*)alloc(2ull * M_ * D_);
    bf16_t* h1   = (bf16_t*)alloc(2ull * M_ * F_);

    pack_embT_kernel<<<4096, 256, 0, stream>>>(emb, embT);
    pack_qkv_kernel <<<2048, 256, 0, stream>>>(Wq, WqP);
    pack_qkv_kernel <<<2048, 256, 0, stream>>>(Wk, WkP);
    pack_qkv_kernel <<<2048, 256, 0, stream>>>(Wv, WvP);
    pack_tiled_kernel<<<2048, 256, 0, stream>>>(Wo, WoP, D_, D_, L_);
    pack_tiled_kernel<<<4096, 256, 0, stream>>>(W1, W1P, D_, F_, L_);
    pack_tiled_kernel<<<4096, 256, 0, stream>>>(W2, W2P, F_, D_, L_);

    embed_kernel<<<M_, 256, 0, stream>>>(X, emb, pos, xbuf);

    dim3 gD(D_ / GN, M_ / GM);     // (6, 16)
    dim3 gF(F_ / GN, M_ / GM);     // (24, 16)
    dim3 gV(V_ / GN, M_ / GM);     // (250, 16)
    dim3 gA(S_ / 16, H_, B_);      // (64, 12, 2)

    for (int l = 0; l < L_; ++l) {
        size_t wDD = (size_t)l * D_ * D_;
        size_t wDF = (size_t)l * D_ * F_;
        layernorm_kernel<<<M_, 256, 0, stream>>>(xbuf, ln1_s + l * D_, ln1_b + l * D_, xn);
        gemm_bf16_kernel<<<gD, 256, 0, stream>>>(xn, WqP + wDD, bq + l * D_, qb_,
                                                 M_, D_, D_, MODE_QKV);
        gemm_bf16_kernel<<<gD, 256, 0, stream>>>(xn, WkP + wDD, bk + l * D_, kb_,
                                                 M_, D_, D_, MODE_QKV);
        gemm_bf16_kernel<<<gD, 256, 0, stream>>>(xn, WvP + wDD, bv + l * D_, vb_,
                                                 M_, D_, D_, MODE_VT);
        attn_kernel<<<gA, 32, 0, stream>>>(qb_, kb_, vb_, atto);
        gemm_bf16_kernel<<<gD, 256, 0, stream>>>(atto, WoP + wDD, bo + l * D_, xbuf,
                                                 M_, D_, D_, MODE_RES_F32);
        layernorm_kernel<<<M_, 256, 0, stream>>>(xbuf, ln2_s + l * D_, ln2_b + l * D_, xn);
        gemm_bf16_kernel<<<gF, 256, 0, stream>>>(xn, W1P + wDF, b1 + l * F_, h1,
                                                 M_, F_, D_, MODE_BF16_RELU);
        gemm_bf16_kernel<<<gD, 256, 0, stream>>>(h1, W2P + wDF, b2 + l * D_, xbuf,
                                                 M_, D_, F_, MODE_RES_F32);
    }

    layernorm_kernel<<<M_, 256, 0, stream>>>(xbuf, lnf_s, lnf_b, xn);
    gemm_bf16_kernel<<<gV, 256, 0, stream>>>(xn, embT, head_b, d_out,
                                             M_, V_, D_, MODE_F32_BIAS);
}
